// GPT_7464653160476
// MI455X (gfx1250) — matrix operand order repaired and checked by reading, not compiled, not run on hardware
//
#include <hip/hip_runtime.h>

// ---------------------------------------------------------------------------
// GPT-2 small forward for gfx1250 (MI455X).
// All matmuls via v_wmma_f32_16x16x32_bf16; GEMM tiles staged in LDS with
// async global->LDS copies (ASYNCcnt) when the toolchain exposes them.
// ---------------------------------------------------------------------------

typedef __bf16 bf16;
typedef __attribute__((ext_vector_type(16))) __bf16 bf16x16;
typedef __attribute__((ext_vector_type(8)))  __bf16 bf16x8;
typedef __attribute__((ext_vector_type(8)))  float  f32x8;
typedef __attribute__((__vector_size__(16))) int    i32x4v;  // builtin's b128 unit

// --- CDNA5 async global->LDS copy (guarded; falls back to load+ds_store) ----
#ifdef __has_builtin
#if __has_builtin(__builtin_amdgcn_global_load_async_to_lds_b128) && \
    __has_builtin(__builtin_amdgcn_s_wait_asynccnt)
#define CDNA5_ASYNC 1
#endif
#endif
#ifndef CDNA5_ASYNC
#define CDNA5_ASYNC 0
#endif

#if CDNA5_ASYNC
// Signature (from hipcc diagnostic): (int4 AS(1)*, int4 AS(3)*, imm offset, imm cpol)
#define ASYNC_CP16(ldsPtr, gPtr)                                                         \
    __builtin_amdgcn_global_load_async_to_lds_b128(                                      \
        (__attribute__((address_space(1))) i32x4v*)(unsigned long long)(gPtr),           \
        (__attribute__((address_space(3))) i32x4v*)(unsigned)(unsigned long long)(ldsPtr), \
        0, 0)
#define ASYNC_WAIT() __builtin_amdgcn_s_wait_asynccnt(0)
#else
#define ASYNC_CP16(ldsPtr, gPtr) (*(bf16x8*)(ldsPtr) = *(const bf16x8*)(gPtr))
#define ASYNC_WAIT()
#endif

// ---------------------------------------------------------------------------
// f32 -> bf16 conversion
// ---------------------------------------------------------------------------
__global__ void cvt_f32_bf16_kernel(const float* __restrict__ in,
                                    bf16* __restrict__ out, long long n) {
    long long i = (long long)blockIdx.x * blockDim.x + threadIdx.x;
    if (i < n) out[i] = (bf16)in[i];
}

// ---------------------------------------------------------------------------
// Embedding: x[b,t,:] = tok_emb[idx[b,t]] + pos_emb[t]
// ---------------------------------------------------------------------------
__global__ void embed_kernel(const int* __restrict__ idx,
                             const float* __restrict__ tok,
                             const float* __restrict__ pos,
                             float* __restrict__ x, int T, int D, long long total) {
    long long i = (long long)blockIdx.x * blockDim.x + threadIdx.x;
    if (i >= total) return;
    int d = (int)(i % D);
    long long bt = i / D;
    int t = (int)(bt % T);
    int token = idx[bt];
    x[i] = tok[(size_t)token * D + d] + pos[(size_t)t * D + d];
}

// ---------------------------------------------------------------------------
// LayerNorm (fp32 in, bf16 out). One 256-thread block per row.
// ---------------------------------------------------------------------------
__global__ void layernorm_kernel(const float* __restrict__ x,
                                 const float* __restrict__ g,
                                 const float* __restrict__ b,
                                 bf16* __restrict__ out, int D) {
    __shared__ float s1[256];
    __shared__ float s2[256];
    const int row = blockIdx.x;
    const float* xr = x + (size_t)row * D;
    float sum = 0.f, sq = 0.f;
    for (int i = threadIdx.x; i < D; i += blockDim.x) {
        float v = xr[i];
        sum += v; sq += v * v;
    }
    s1[threadIdx.x] = sum; s2[threadIdx.x] = sq;
    __syncthreads();
    for (int s = 128; s > 0; s >>= 1) {
        if ((int)threadIdx.x < s) {
            s1[threadIdx.x] += s1[threadIdx.x + s];
            s2[threadIdx.x] += s2[threadIdx.x + s];
        }
        __syncthreads();
    }
    const float mu  = s1[0] / D;
    const float var = s2[0] / D - mu * mu;
    const float inv = rsqrtf(var + 1e-5f);
    for (int i = threadIdx.x; i < D; i += blockDim.x) {
        float v = (xr[i] - mu) * inv * g[i] + b[i];
        out[(size_t)row * D + i] = (bf16)v;
    }
}

// ---------------------------------------------------------------------------
// LDS-tiled WMMA GEMM:
//   C[M,N](f32) = act( A[M,K](bf16) @ B[N,K](bf16)^T + bias ) + residual
//
// Block = 256 threads (8 waves). Block tile: BM=128, BN=64, BK=32.
// Waves laid out 4(M) x 2(N); each wave owns a 32x32 output region =
// 2x2 grid of 16x16 WMMA accumulators (A frags reused across N, B across M).
// Tiles are staged in LDS, double-buffered; fills use async global->LDS
// (ASYNCcnt) so the next tile streams in while the current one is computed.
// Per K-step: 12KB global traffic for 512 KFLOP (~42 FLOP/B).
//
// Fragment layouts (ISA 7.12.2, wave32):
//   A 16x32 bf16: lane L -> row M=L%16; halves 0..7  K=8*(L/16)+i,
//                                        halves 8..15 K=16+8*(L/16)+(i-8)
//   B 32x16 bf16: lane L -> col N=L%16; halves 0..7  K=8*(L/16)+i, halves 8..15 K=16+8*(L/16)+(i-8)
//   C 16x16 f32 : VGPR j, lane L -> (M=j+8*(L/16), N=L%16)
// ---------------------------------------------------------------------------
#define BM 128
#define BN 64
#define BK 32

template <bool GELU>
__global__ void __launch_bounds__(256)
wmma_gemm_kernel(const bf16* __restrict__ A, long long sA,
                 const bf16* __restrict__ Bw, long long sB,
                 float* __restrict__ C, long long sC,
                 const float* __restrict__ bias,
                 const float* __restrict__ residual,
                 int M, int N, int K,
                 int lda, int ldb, int ldc) {
    __shared__ __attribute__((aligned(16))) bf16 As[2][BM * BK]; // 16 KB
    __shared__ __attribute__((aligned(16))) bf16 Bs[2][BN * BK]; //  8 KB

    const int tid  = threadIdx.x;
    const int wave = tid >> 5;
    const int lane = tid & 31;
    const int l16  = lane & 15;
    const int lhi  = lane >> 4;

    const int tileM0 = blockIdx.y * BM;   // M always a multiple of BM here
    const int tileN0 = blockIdx.x * BN;
    const int wm = (wave >> 1) * 32;      // wave's M offset within block tile
    const int wn = (wave & 1) * 32;       // wave's N offset within block tile

    const long long bz = blockIdx.z;
    const bf16* Ab = A  + bz * sA;
    const bf16* Bb = Bw + bz * sB;
    float*      Cb = C  + bz * sC;

    // ---- cooperative tile fill: 16B chunk per issue ----
    auto fill = [&](int buf, int k0) {
        // A tile: BM x BK halves = 512 chunks of 8 halves; 2 per thread
#pragma unroll
        for (int i = 0; i < 2; ++i) {
            const int c   = tid + 256 * i;      // 0..511
            const int row = c >> 2;             // 0..127
            const int off = (c & 3) * 8;        // 0,8,16,24
            const bf16* gp = Ab + (size_t)(tileM0 + row) * lda + k0 + off;
            ASYNC_CP16(&As[buf][row * BK + off], gp);
        }
        // B tile: BN x BK halves = 256 chunks; 1 per thread
        {
            const int row = tid >> 2;           // 0..63
            const int off = (tid & 3) * 8;
            const int n   = tileN0 + row;
            const int nc  = (n < N) ? n : (N - 1);  // clamp ragged N for loads
            const bf16* gp = Bb + (size_t)nc * ldb + k0 + off;
            ASYNC_CP16(&Bs[buf][row * BK + off], gp);
        }
    };

    f32x8 acc[2][2] = {};

    const int nk = K / BK;
    fill(0, 0);
    ASYNC_WAIT();
    __syncthreads();

    for (int kt = 0; kt < nk; ++kt) {
        const int cur = kt & 1;
        if (kt + 1 < nk) fill(1 - cur, (kt + 1) * BK);  // overlap next tile

        // load fragments from LDS
        bf16x16 afrag[2], bfrag[2];
#pragma unroll
        for (int i = 0; i < 2; ++i) {
            const bf16* p = &As[cur][(wm + 16 * i + l16) * BK + 8 * lhi];
            bf16x8 lo = *(const bf16x8*)p;
            bf16x8 hi = *(const bf16x8*)(p + 16);
#pragma unroll
            for (int e = 0; e < 8; ++e) { afrag[i][e] = lo[e]; afrag[i][e + 8] = hi[e]; }
        }
#pragma unroll
        for (int j = 0; j < 2; ++j) {
            const bf16* p = &Bs[cur][(wn + 16 * j + l16) * BK + 8 * lhi];
            bf16x8 lo = *(const bf16x8*)p;
            bf16x8 hi = *(const bf16x8*)(p + 16);
#pragma unroll
            for (int e = 0; e < 8; ++e) { bfrag[j][e] = lo[e]; bfrag[j][e + 8] = hi[e]; }
        }

#pragma unroll
        for (int i = 0; i < 2; ++i)
#pragma unroll
            for (int j = 0; j < 2; ++j)
                acc[i][j] = __builtin_amdgcn_wmma_f32_16x16x32_bf16(
                    false, afrag[i], false, bfrag[j],
                    (short)0, acc[i][j], false, false);

        ASYNC_WAIT();       // this wave's async fills of the other buffer done
        __syncthreads();    // all waves done reading `cur` + fills visible
    }

    // ---- epilogue: bias, optional exact GELU, optional residual ----
#pragma unroll
    for (int i = 0; i < 2; ++i) {
#pragma unroll
        for (int j = 0; j < 2; ++j) {
            const int n = tileN0 + wn + 16 * j + l16;
            if (n < N) {
                const float bn = (bias != nullptr) ? bias[n] : 0.f;
#pragma unroll
                for (int v = 0; v < 8; ++v) {
                    const int m = tileM0 + wm + 16 * i + v + 8 * lhi;
                    float val = acc[i][j][v] + bn;
                    if (GELU) val = 0.5f * val * (1.f + erff(val * 0.70710678118654752f));
                    const size_t ci = (size_t)m * ldc + n;
                    if (residual != nullptr) val += residual[ci];
                    Cb[ci] = val;
                }
            }
        }
    }
}

// ---------------------------------------------------------------------------
// Split QKV: qkv[BT,3D] fp32 -> per-head bf16 Q[bh,T,HD], K[bh,T,HD], V^T[bh,HD,T]
// ---------------------------------------------------------------------------
__global__ void split_qkv_kernel(const float* __restrict__ qkv,
                                 bf16* __restrict__ q, bf16* __restrict__ k,
                                 bf16* __restrict__ vT,
                                 int T, int H, int HD, int D, long long total) {
    long long i = (long long)blockIdx.x * blockDim.x + threadIdx.x;
    if (i >= total) return;
    const int d  = (int)(i % HD);
    const int t  = (int)((i / HD) % T);
    const long long bh = i / ((long long)HD * T);
    const int b = (int)(bh / H);
    const int h = (int)(bh % H);
    const float* src = qkv + ((size_t)(b * T + t) * (3 * D)) + h * HD + d;
    q[i] = (bf16)src[0];
    k[i] = (bf16)src[D];
    vT[((size_t)bh * HD + d) * T + t] = (bf16)src[2 * D];
}

// ---------------------------------------------------------------------------
// Causal softmax (applies scale); writes bf16 probabilities, padded with 0.
// ---------------------------------------------------------------------------
__global__ void softmax_causal_kernel(const float* __restrict__ att,
                                      bf16* __restrict__ attp,
                                      int T, float scale) {
    __shared__ float red[256];
    const long long r  = blockIdx.x;
    const long long bh = r / T;
    const int       q  = (int)(r % T);
    const float* row  = att  + bh * (long long)T * T + (long long)q * T;
    bf16*        prow = attp + bh * (long long)T * T + (long long)q * T;
    const int nvalid = q + 1;

    float mx = -3.0e38f;
    for (int j = threadIdx.x; j < nvalid; j += blockDim.x)
        mx = fmaxf(mx, row[j] * scale);
    red[threadIdx.x] = mx;
    __syncthreads();
    for (int s = 128; s > 0; s >>= 1) {
        if ((int)threadIdx.x < s)
            red[threadIdx.x] = fmaxf(red[threadIdx.x], red[threadIdx.x + s]);
        __syncthreads();
    }
    mx = red[0];
    __syncthreads();

    float sum = 0.f;
    for (int j = threadIdx.x; j < nvalid; j += blockDim.x)
        sum += expf(row[j] * scale - mx);
    red[threadIdx.x] = sum;
    __syncthreads();
    for (int s = 128; s > 0; s >>= 1) {
        if ((int)threadIdx.x < s) red[threadIdx.x] += red[threadIdx.x + s];
        __syncthreads();
    }
    const float inv = 1.f / red[0];

    for (int j = threadIdx.x; j < T; j += blockDim.x) {
        const float p = (j < nvalid) ? expf(row[j] * scale - mx) * inv : 0.f;
        prow[j] = (bf16)p;
    }
}

// ---------------------------------------------------------------------------
// Merge heads: y[bh,T,HD] fp32 -> yb[BT,D] bf16
// ---------------------------------------------------------------------------
__global__ void merge_heads_kernel(const float* __restrict__ y,
                                   bf16* __restrict__ yb,
                                   int T, int H, int HD, int D, long long total) {
    long long i = (long long)blockIdx.x * blockDim.x + threadIdx.x;
    if (i >= total) return;
    const int c = (int)(i % D);
    const int h = c / HD;
    const int d = c % HD;
    const long long bt = i / D;
    const int t = (int)(bt % T);
    const int b = (int)(bt / T);
    yb[i] = (bf16)y[(((size_t)(b * H + h) * T + t) * HD) + d];
}

// ---------------------------------------------------------------------------
// Host-side orchestration
// ---------------------------------------------------------------------------
extern "C" void kernel_launch(void* const* d_in, const int* in_sizes, int n_in,
                              void* d_out, int out_size, void* d_ws, size_t ws_size,
                              hipStream_t stream) {
    (void)in_sizes; (void)n_in; (void)out_size; (void)ws_size;

    const int Bb = 2, T = 1024, D = 768, H = 12, HD = 64, L = 12, V = 50257;
    const int BT = Bb * T;            // 2048
    const int D3 = 3 * D, D4 = 4 * D; // 2304, 3072
    const float SCALE = 0.125f;       // HD^-0.5

    const int*   idx   = (const int*)d_in[0];
    const float* tok   = (const float*)d_in[1];
    const float* pos   = (const float*)d_in[2];
    const float* wqkv  = (const float*)d_in[3];
    const float* bqkv  = (const float*)d_in[4];
    const float* wproj = (const float*)d_in[5];
    const float* bproj = (const float*)d_in[6];
    const float* ln1g  = (const float*)d_in[7];
    const float* ln1b  = (const float*)d_in[8];
    const float* ln2g  = (const float*)d_in[9];
    const float* ln2b  = (const float*)d_in[10];
    const float* w1    = (const float*)d_in[11];
    const float* b1    = (const float*)d_in[12];
    const float* w2    = (const float*)d_in[13];
    const float* b2    = (const float*)d_in[14];
    const float* lnfg  = (const float*)d_in[15];
    const float* lnfb  = (const float*)d_in[16];
    float* out = (float*)d_out;

    // Workspace bump allocator (256B aligned)
    size_t off = 0;
    auto alloc = [&](size_t bytes) -> void* {
        void* p = (char*)d_ws + off;
        off += (bytes + 255) & ~(size_t)255;
        return p;
    };

    bf16* wqkv_bf  = (bf16*)alloc(sizeof(bf16) * (size_t)L * D3 * D);
    bf16* wproj_bf = (bf16*)alloc(sizeof(bf16) * (size_t)L * D * D);
    bf16* w1_bf    = (bf16*)alloc(sizeof(bf16) * (size_t)L * D4 * D);
    bf16* w2_bf    = (bf16*)alloc(sizeof(bf16) * (size_t)L * D * D4);
    bf16* emb_bf   = (bf16*)alloc(sizeof(bf16) * (size_t)V * D);
    float* x    = (float*)alloc(sizeof(float) * (size_t)BT * D);
    bf16*  h    = (bf16*) alloc(sizeof(bf16)  * (size_t)BT * D);
    float* qkv  = (float*)alloc(sizeof(float) * (size_t)BT * D3);
    bf16*  qb   = (bf16*) alloc(sizeof(bf16)  * (size_t)BT * D);
    bf16*  kb   = (bf16*) alloc(sizeof(bf16)  * (size_t)BT * D);
    bf16*  vT   = (bf16*) alloc(sizeof(bf16)  * (size_t)BT * D);
    float* att  = (float*)alloc(sizeof(float) * (size_t)Bb * H * T * T);
    bf16*  attp = (bf16*) alloc(sizeof(bf16)  * (size_t)Bb * H * T * T);
    float* y    = (float*)alloc(sizeof(float) * (size_t)BT * D);
    bf16*  yb   = (bf16*) alloc(sizeof(bf16)  * (size_t)BT * D);
    float* ff1  = (float*)alloc(sizeof(float) * (size_t)BT * D4);
    bf16*  ff1b = (bf16*) alloc(sizeof(bf16)  * (size_t)BT * D4);

    auto cvt = [&](const float* src, bf16* dst, long long n) {
        const unsigned blocks = (unsigned)((n + 255) / 256);
        cvt_f32_bf16_kernel<<<dim3(blocks), dim3(256), 0, stream>>>(src, dst, n);
    };

    auto gemm = [&](const bf16* A, long long sA, int lda,
                    const bf16* Bm, long long sB, int ldb,
                    float* C, long long sC, int ldc,
                    const float* bias, const float* res,
                    int M, int N, int K, int batch, bool gelu) {
        dim3 grid((unsigned)((N + BN - 1) / BN), (unsigned)(M / BM), (unsigned)batch);
        dim3 block(256);
        if (gelu)
            wmma_gemm_kernel<true><<<grid, block, 0, stream>>>(
                A, sA, Bm, sB, C, sC, bias, res, M, N, K, lda, ldb, ldc);
        else
            wmma_gemm_kernel<false><<<grid, block, 0, stream>>>(
                A, sA, Bm, sB, C, sC, bias, res, M, N, K, lda, ldb, ldc);
    };

    // ---- Weight conversions (fp32 -> bf16), once per call ----
    cvt(wqkv, wqkv_bf, (long long)L * D3 * D);
    cvt(wproj, wproj_bf, (long long)L * D * D);
    cvt(w1, w1_bf, (long long)L * D4 * D);
    cvt(w2, w2_bf, (long long)L * D * D4);
    cvt(tok, emb_bf, (long long)V * D);

    // ---- Embedding ----
    {
        const long long total = (long long)BT * D;
        embed_kernel<<<dim3((unsigned)((total + 255) / 256)), dim3(256), 0, stream>>>(
            idx, tok, pos, x, T, D, total);
    }

    const long long perHead   = (long long)T * HD;   // 65536
    const long long perHeadSq = (long long)T * T;    // 1048576
    const long long nHeadEl   = (long long)BT * D;   // 1572864

    for (int l = 0; l < L; ++l) {
        const bf16*  Wq = wqkv_bf  + (size_t)l * D3 * D;
        const bf16*  Wp = wproj_bf + (size_t)l * D * D;
        const bf16*  W1 = w1_bf    + (size_t)l * D4 * D;
        const bf16*  W2 = w2_bf    + (size_t)l * D * D4;
        const float* Bq = bqkv  + (size_t)l * D3;
        const float* Bp = bproj + (size_t)l * D;
        const float* B1 = b1    + (size_t)l * D4;
        const float* B2 = b2    + (size_t)l * D;

        // LN1: x -> h (bf16)
        layernorm_kernel<<<dim3(BT), dim3(256), 0, stream>>>(
            x, ln1g + (size_t)l * D, ln1b + (size_t)l * D, h, D);

        // QKV GEMM: [2048,768] x [2304,768]^T + bias -> qkv fp32
        gemm(h, 0, D, Wq, 0, D, qkv, 0, D3, Bq, nullptr, BT, D3, D, 1, false);

        // Split into per-head bf16 Q, K, V^T
        split_qkv_kernel<<<dim3((unsigned)((nHeadEl + 255) / 256)), dim3(256), 0, stream>>>(
            qkv, qb, kb, vT, T, H, HD, D, nHeadEl);

        // Scores: per (b,h): Q[1024,64] x K[1024,64]^T -> att fp32 [1024,1024]
        gemm(qb, perHead, HD, kb, perHead, HD, att, perHeadSq, T,
             nullptr, nullptr, T, T, HD, Bb * H, false);

        // Causal softmax (applies SCALE), writes bf16 probabilities
        softmax_causal_kernel<<<dim3((unsigned)(Bb * H * T)), dim3(256), 0, stream>>>(
            att, attp, T, SCALE);

        // att @ V: per (b,h): P[1024,1024] x (V^T)[64,1024]^T -> y fp32 [1024,64]
        gemm(attp, perHeadSq, T, vT, perHead, T, y, perHead, HD,
             nullptr, nullptr, T, HD, T, Bb * H, false);

        // Merge heads -> yb bf16 [2048,768]
        merge_heads_kernel<<<dim3((unsigned)((nHeadEl + 255) / 256)), dim3(256), 0, stream>>>(
            y, yb, T, H, HD, D, nHeadEl);

        // Proj + residual: x = x + yb @ Wp^T + Bp (in-place residual)
        gemm(yb, 0, D, Wp, 0, D, x, 0, D, Bp, x, BT, D, D, 1, false);

        // LN2: x -> h (bf16)
        layernorm_kernel<<<dim3(BT), dim3(256), 0, stream>>>(
            x, ln2g + (size_t)l * D, ln2b + (size_t)l * D, h, D);

        // FC1 + exact GELU -> ff1 fp32 [2048,3072]
        gemm(h, 0, D, W1, 0, D, ff1, 0, D4, B1, nullptr, BT, D4, D, 1, true);

        // ff1 -> bf16
        cvt(ff1, ff1b, (long long)BT * D4);

        // FC2 + residual: x = x + ff1b @ W2^T + B2
        gemm(ff1b, 0, D4, W2, 0, D4, x, 0, D, B2, x, BT, D, D4, 1, false);
    }

    // Final LN: x -> h (bf16)
    layernorm_kernel<<<dim3(BT), dim3(256), 0, stream>>>(x, lnfg, lnfb, h, D);

    // LM head (tied embeddings): [2048,768] x [50257,768]^T -> out fp32
    gemm(h, 0, D, emb_bf, 0, D, out, 0, V, nullptr, nullptr, BT, V, D, 1, false);
}
